// SelfAttention_68058051773050
// MI455X (gfx1250) — matrix operand order repaired and checked by reading, not compiled, not run on hardware
//
#include <hip/hip_runtime.h>
#include <hip/hip_bf16.h>

// CDNA5 (gfx1250) wave32 WMMA flash-attention for SAGAN self-attention.
//   K1: fused QKV projection (x[f32] @ {wq,wk,wv}^T + bias)
//       -> qk[b][n][64] bf16 (q cols 0..31, k cols 32..63)   [2 MB]
//       -> vT[b][c][n] bf16 (channel-major, keys contiguous) [8 MB]
//   K2: flash attention (online softmax) + gamma*out + x epilogue -> out[f32]
// All K2 WMMA fragments are built from contiguous 16-byte chunks (b128 loads).

typedef __attribute__((ext_vector_type(16))) __bf16 v16bf;
typedef __attribute__((ext_vector_type(8)))  __bf16 v8bf;
typedef __attribute__((ext_vector_type(8)))  float  v8f;

#define BATCH 4
#define CIN   256
#define NPIX  4096
#define QKW   64       // packed q|k row width per pixel
#define PPAD  40       // LDS P row stride (bf16): 80 B keeps 16B align, distinct banks

// K1's A/B element->K mapping, contiguous k = 16h + j (lane half h, vector element j)
__device__ __forceinline__ int a_kidx(int j, int h) {
    return 16 * h + j;
}

__device__ __forceinline__ v16bf join16(v8bf lo, v8bf hi) {
    v16bf r;
    #pragma unroll
    for (int j = 0; j < 8; ++j) { r[j] = lo[j]; r[j + 8] = hi[j]; }
    return r;
}

// ---------------------------------------------------------------------------
// Kernel 1: fused QKV projection. One wave = one 16(pixel) x 16(outch) tile,
// K-loop over CIN in steps of 32 (8 WMMAs). oT 0..1 -> q, 2..3 -> k, 4..19 -> v.
// ---------------------------------------------------------------------------
__global__ void __launch_bounds__(256)
qkv_proj_kernel(const float* __restrict__ x,
                const float* __restrict__ wq, const float* __restrict__ bq,
                const float* __restrict__ wk, const float* __restrict__ bk,
                const float* __restrict__ wv, const float* __restrict__ bv,
                __bf16* __restrict__ qk, __bf16* __restrict__ vbuf)
{
    const int lane = threadIdx.x & 31;
    const int h    = lane >> 4;
    const int col  = lane & 15;
    const int wave = threadIdx.x >> 5;

    int task = blockIdx.x * 8 + wave;            // BATCH*256*20 = 20480 tasks
    const int oT = task % 20;  task /= 20;
    const int nT = task % 256; task /= 256;
    const int b  = task;

    const float* w; const float* bias; int olocal;
    if (oT < 2)      { w = wq; bias = bq; olocal = oT * 16; }
    else if (oT < 4) { w = wk; bias = bk; olocal = (oT - 2) * 16; }
    else             { w = wv; bias = bv; olocal = (oT - 4) * 16; }

    const int n0 = nT * 16;
    v8f acc = {};

    for (int cb = 0; cb < CIN; cb += 32) {
        // A: 16 pixels (M = lane&15) x 32 in-channels (K), gathered from x[b][c][n]
        v16bf afrag;
        #pragma unroll
        for (int j = 0; j < 16; ++j) {
            int c = cb + a_kidx(j, h);
            afrag[j] = (__bf16)x[((size_t)b * CIN + c) * NPIX + n0 + col];
        }
        // B: 32 in-channels (K = 16h + j) x 16 out-channels; contiguous f32 row.
        const float4* wr = (const float4*)(w + (size_t)(olocal + col) * CIN + cb + h * 16);
        float4 w0 = wr[0], w1 = wr[1], w2 = wr[2], w3 = wr[3];
        v16bf bfrag;
        bfrag[0]=(__bf16)w0.x; bfrag[1]=(__bf16)w0.y; bfrag[2]=(__bf16)w0.z; bfrag[3]=(__bf16)w0.w;
        bfrag[4]=(__bf16)w1.x; bfrag[5]=(__bf16)w1.y; bfrag[6]=(__bf16)w1.z; bfrag[7]=(__bf16)w1.w;
        bfrag[8]=(__bf16)w2.x; bfrag[9]=(__bf16)w2.y; bfrag[10]=(__bf16)w2.z; bfrag[11]=(__bf16)w2.w;
        bfrag[12]=(__bf16)w3.x; bfrag[13]=(__bf16)w3.y; bfrag[14]=(__bf16)w3.z; bfrag[15]=(__bf16)w3.w;

        acc = __builtin_amdgcn_wmma_f32_16x16x32_bf16(false, afrag, false, bfrag,
                                                      (short)0, acc, false, false);
    }

    const float bval = bias[olocal + col];
    if (oT < 4) {
        // q/k: rows (pixels) strided by QKW in the packed buffer.
        #pragma unroll
        for (int r = 0; r < 8; ++r) {
            int n = n0 + r + 8 * h;
            qk[((size_t)b * NPIX + n) * QKW + oT * 16 + col] = (__bf16)(acc[r] + bval);
        }
    } else {
        // v: channel-major; the 8 rows per lane are 8 consecutive pixels -> 1 b128 store.
        int c = (oT - 4) * 16 + col;
        v8bf pack;
        #pragma unroll
        for (int r = 0; r < 8; ++r) pack[r] = (__bf16)(acc[r] + bval);
        *(v8bf*)(vbuf + ((size_t)b * CIN + c) * NPIX + n0 + 8 * h) = pack;
    }
}

// ---------------------------------------------------------------------------
// Kernel 2: flash attention. 256 threads = 8 waves; wave pair (2w, 2w+1) owns
// one 16-query tile (4 tiles / 64 queries per block); within a pair each wave
// owns 128 output channels (8 accum tiles). Per 32-key block per wave:
// 2 S-WMMAs + online softmax + LDS relayout + 8 PV-WMMAs.
// ---------------------------------------------------------------------------
__global__ void __launch_bounds__(256)
flash_attn_kernel(const __bf16* __restrict__ qk,
                  const __bf16* __restrict__ vbuf,
                  const float*  __restrict__ x,
                  const float*  __restrict__ gamma,
                  float*        __restrict__ out)
{
    __shared__ __bf16 pbuf[8][16][PPAD];         // per-wave P relayout buffer

    const int lane  = threadIdx.x & 31;
    const int h     = lane >> 4;
    const int col   = lane & 15;
    const int wave  = threadIdx.x >> 5;
    const int qt    = wave >> 1;                 // query tile in block (0..3)
    const int chh   = (wave & 1) * 128;          // channel half base
    const int b     = blockIdx.x >> 6;           // 64 blocks per batch
    const int q0    = ((blockIdx.x & 63) * 4 + qt) * 16;

    const __bf16* qkb = qk   + (size_t)b * NPIX * QKW;
    const __bf16* vb  = vbuf + (size_t)b * CIN * NPIX;

    // Q fragment: A-layout 16x32; row is 128B, chunks at elems {8h, 16+8h}.
    const v8bf* qrow = (const v8bf*)(qkb + (size_t)(q0 + col) * QKW);
    const v16bf qa = join16(qrow[h], qrow[2 + h]);

    v8f acc[8] = {v8f{}, v8f{}, v8f{}, v8f{}, v8f{}, v8f{}, v8f{}, v8f{}};
    float mrow[8], lrow[8];
    #pragma unroll
    for (int r = 0; r < 8; ++r) { mrow[r] = -INFINITY; lrow[r] = 0.f; }

    for (int kb = 0; kb < NPIX; kb += 32) {
        // K fragments: B-layout 32(Cr) x 16(keys); k channels at elems [32,64).
        const v8bf* kr0 = (const v8bf*)(qkb + (size_t)(kb      + col) * QKW);
        const v8bf* kr1 = (const v8bf*)(qkb + (size_t)(kb + 16 + col) * QKW);
        v16bf kf0 = join16(kr0[4 + h], kr0[6 + h]);
        v16bf kf1 = join16(kr1[4 + h], kr1[6 + h]);

        v8f z = {};
        v8f s0 = __builtin_amdgcn_wmma_f32_16x16x32_bf16(false, qa, false, kf0,
                                                         (short)0, z, false, false);
        v8f s1 = __builtin_amdgcn_wmma_f32_16x16x32_bf16(false, qa, false, kf1,
                                                         (short)0, z, false, false);

        // Online softmax; row (r + 8h) lives across the 16 lanes of one half.
        float scale[8];
        #pragma unroll
        for (int r = 0; r < 8; ++r) {
            float vmax = fmaxf(s0[r], s1[r]);
            #pragma unroll
            for (int m = 8; m >= 1; m >>= 1)
                vmax = fmaxf(vmax, __shfl_xor(vmax, m, 32));
            float newm = fmaxf(mrow[r], vmax);
            scale[r] = __expf(mrow[r] - newm);
            mrow[r] = newm;
            float p0 = __expf(s0[r] - newm);
            float p1 = __expf(s1[r] - newm);
            float psum = p0 + p1;
            #pragma unroll
            for (int m = 8; m >= 1; m >>= 1)
                psum += __shfl_xor(psum, m, 32);
            lrow[r] = lrow[r] * scale[r] + psum;
            pbuf[wave][r + 8 * h][col]      = (__bf16)p0;   // D-layout -> row-major tile
            pbuf[wave][r + 8 * h][16 + col] = (__bf16)p1;
        }
        #pragma unroll
        for (int t = 0; t < 8; ++t)
            #pragma unroll
            for (int r = 0; r < 8; ++r)
                acc[t][r] *= scale[r];

        // Intra-wave cross-lane LDS dependency: drain DS before re-reading.
        asm volatile("s_wait_dscnt 0" ::: "memory");

        // P as A-layout 16x32 fragment: two ds b128 chunks per lane.
        const v8bf* prow = (const v8bf*)(&pbuf[wave][col][0]);
        v16bf pf = join16(prow[h], prow[2 + h]);

        // P @ V: 8 channel tiles; vT keys are contiguous -> two b128 per tile.
        #pragma unroll
        for (int t = 0; t < 8; ++t) {
            const v8bf* vrow = (const v8bf*)(vb + (size_t)(chh + t * 16 + col) * NPIX
                                                + kb);
            v16bf vf = join16(vrow[h], vrow[2 + h]);
            acc[t] = __builtin_amdgcn_wmma_f32_16x16x32_bf16(false, pf, false, vf,
                                                             (short)0, acc[t], false, false);
        }
    }

    // Epilogue: out = gamma * (acc / l) + x; 8 consecutive pixels per lane -> float4 x2.
    const float g = gamma[0];
    #pragma unroll
    for (int t = 0; t < 8; ++t) {
        int c = chh + t * 16 + col;
        size_t rowbase = ((size_t)b * CIN + c) * NPIX + q0 + 8 * h;
        const float4* xr = (const float4*)(x + rowbase);
        float4 x0 = xr[0], x1 = xr[1];
        float4 o0, o1;
        o0.x = g * (acc[t][0] / lrow[0]) + x0.x;
        o0.y = g * (acc[t][1] / lrow[1]) + x0.y;
        o0.z = g * (acc[t][2] / lrow[2]) + x0.z;
        o0.w = g * (acc[t][3] / lrow[3]) + x0.w;
        o1.x = g * (acc[t][4] / lrow[4]) + x1.x;
        o1.y = g * (acc[t][5] / lrow[5]) + x1.y;
        o1.z = g * (acc[t][6] / lrow[6]) + x1.z;
        o1.w = g * (acc[t][7] / lrow[7]) + x1.w;
        float4* orow = (float4*)(out + rowbase);
        orow[0] = o0;
        orow[1] = o1;
    }
}

extern "C" void kernel_launch(void* const* d_in, const int* in_sizes, int n_in,
                              void* d_out, int out_size, void* d_ws, size_t ws_size,
                              hipStream_t stream) {
    (void)in_sizes; (void)n_in; (void)out_size; (void)ws_size;
    const float* x     = (const float*)d_in[0];
    const float* wq    = (const float*)d_in[1];
    const float* bq    = (const float*)d_in[2];
    const float* wk    = (const float*)d_in[3];
    const float* bk    = (const float*)d_in[4];
    const float* wv    = (const float*)d_in[5];
    const float* bv    = (const float*)d_in[6];
    const float* gamma = (const float*)d_in[7];
    float* out = (float*)d_out;

    __bf16* qkbuf = (__bf16*)d_ws;                                   // 2 MB
    __bf16* vbufp = (__bf16*)d_ws + (size_t)BATCH * NPIX * QKW;      // 8 MB

    qkv_proj_kernel<<<2560, 256, 0, stream>>>(x, wq, bq, wk, bk, wv, bv, qkbuf, vbufp);
    flash_attn_kernel<<<BATCH * (NPIX / 64), 256, 0, stream>>>(qkbuf, vbufp, x, gamma, out);
}
